// FP8Linear_42812234006536
// MI455X (gfx1250) — 8-bit matrix operand order repaired and checked by reading, compiled, not run on hardware
//
#include <hip/hip_runtime.h>
#include <hip/hip_bf16.h>

typedef __attribute__((ext_vector_type(16))) int   v16i;
typedef __attribute__((ext_vector_type(8)))  float v8f;
typedef int v4i32 __attribute__((vector_size(16)));

#define FP8_MAX 448.0f
#define BLK 128

#define AS1 __attribute__((address_space(1)))
#define AS3 __attribute__((address_space(3)))

#if __has_builtin(__builtin_amdgcn_global_load_async_to_lds_b128)
#define USE_ASYNC_LDS 1
#else
#define USE_ASYNC_LDS 0
#endif

static __device__ __forceinline__ AS1 v4i32* as_g(const void* p) {
    return (AS1 v4i32*)(unsigned long long)p;
}
static __device__ __forceinline__ AS3 v4i32* as_l(void* p) {
    return (AS3 v4i32*)(unsigned)(unsigned long long)p;
}
static __device__ __forceinline__ void wait_async0() {
#if __has_builtin(__builtin_amdgcn_s_wait_asynccnt)
    __builtin_amdgcn_s_wait_asynccnt(0);
#else
    asm volatile("s_wait_asynccnt 0" ::: "memory");
#endif
}

#if __has_builtin(__builtin_amdgcn_cvt_pk_fp8_f32)
static __device__ __forceinline__ unsigned int pack4_fp8(float a, float b, float c, float d) {
    int lo  = __builtin_amdgcn_cvt_pk_fp8_f32(a, b, 0, false);
    int all = __builtin_amdgcn_cvt_pk_fp8_f32(c, d, lo, true);
    return (unsigned int)all;
}
#else
static __device__ __forceinline__ unsigned char f32_to_e4m3_sw(float f) {
    unsigned int u    = __float_as_uint(f);
    unsigned int sign = (u >> 24) & 0x80u;
    float af = fabsf(f);
    if (af > FP8_MAX) af = FP8_MAX;
    if (af == 0.0f) return (unsigned char)sign;
    int e = (int)(__float_as_uint(af) >> 23) - 127;
    unsigned int out;
    if (e < -6) {
        int r = (int)rintf(af * 512.0f);
        out = (unsigned int)r;
    } else {
        int r = (int)rintf(ldexpf(af, 3 - e));
        if (r == 16) { r = 8; e += 1; }
        if (e > 8) out = 0x7Eu;
        else out = (unsigned int)(((e + 7) << 3) | (r & 7));
    }
    return (unsigned char)(out | sign);
}
static __device__ __forceinline__ unsigned int pack4_fp8(float a, float b, float c, float d) {
    return (unsigned int)f32_to_e4m3_sw(a)
         | ((unsigned int)f32_to_e4m3_sw(b) << 8)
         | ((unsigned int)f32_to_e4m3_sw(c) << 16)
         | ((unsigned int)f32_to_e4m3_sw(d) << 24);
}
#endif

static __device__ __forceinline__ float clamp448(float v) {
    return fminf(fmaxf(v, -FP8_MAX), FP8_MAX);
}

__global__ __launch_bounds__(256)
void quantize_x_kernel(const float* __restrict__ x,
                       unsigned char* __restrict__ xq,
                       float* __restrict__ xs,
                       int groups) {
    int wave = blockIdx.x * (blockDim.x >> 5) + (threadIdx.x >> 5);
    int lane = threadIdx.x & 31;
    if (wave >= groups) return;

    long base = (long)wave * BLK;
    const float4 v = *(const float4*)(x + base + lane * 4);

    float m = fmaxf(fmaxf(fabsf(v.x), fabsf(v.y)), fmaxf(fabsf(v.z), fabsf(v.w)));
    #pragma unroll
    for (int s = 16; s >= 1; s >>= 1)
        m = fmaxf(m, __shfl_xor(m, s, 32));

    m = fmaxf(m, 1e-12f);
    float scale = FP8_MAX / m;

    unsigned int packed = pack4_fp8(clamp448(v.x * scale), clamp448(v.y * scale),
                                    clamp448(v.z * scale), clamp448(v.w * scale));
    *(unsigned int*)(xq + base + lane * 4) = packed;
    if (lane == 0) xs[wave] = 1.0f / scale;
}

__global__ __launch_bounds__(256)
void quantize_w_kernel(const float* __restrict__ w,
                       unsigned char* __restrict__ wq,
                       long n4) {
    long i = (long)blockIdx.x * blockDim.x + threadIdx.x;
    if (i >= n4) return;
    const float4 v = *(const float4*)(w + i * 4);
    *(unsigned int*)(wq + i * 4) =
        pack4_fp8(clamp448(v.x), clamp448(v.y), clamp448(v.z), clamp448(v.w));
}

__global__ __launch_bounds__(256)
void fp8_gemm_kernel(const unsigned char* __restrict__ xq,
                     const unsigned char* __restrict__ wq,
                     const float* __restrict__ xs,
                     const float* __restrict__ wscale,
                     const float* __restrict__ bias,
                     float* __restrict__ out,
                     int M, int N, int K) {
    const int J = K >> 7;
    __shared__ __align__(16) unsigned char As[2][128 * 128];
    __shared__ __align__(16) unsigned char Bs[2][128 * 128];
    __shared__ __align__(16) float         Ss[16 * 128];

    const int tid   = threadIdx.x;
    const int lane  = tid & 31;
    const int wave  = tid >> 5;
    const int waveM = wave & 1;
    const int waveN = wave >> 1;
    const int hi    = lane >> 4;
    const int lm    = lane & 15;

    const long mBase = (long)blockIdx.y * 128;
    const long nBase = (long)blockIdx.x * 128;

    {
        #pragma unroll
        for (int t = 0; t < 2; ++t) {
            int i  = tid + t * 256;
            int r  = i / (J >> 2);
            int c4 = i - r * (J >> 2);
            int j  = c4 << 2;
            float4 v  = ((const float4*)(xs + (mBase + r) * J))[c4];
            float4 ws = ((const float4*)(wscale + (long)blockIdx.x * J))[c4];
            Ss[(j + 0) * 128 + r] = v.x * ws.x;
            Ss[(j + 1) * 128 + r] = v.y * ws.y;
            Ss[(j + 2) * 128 + r] = v.z * ws.z;
            Ss[(j + 3) * 128 + r] = v.w * ws.w;
        }
    }

    float acc[4][2][8];
    #pragma unroll
    for (int a = 0; a < 4; ++a)
        #pragma unroll
        for (int b = 0; b < 2; ++b)
            #pragma unroll
            for (int e = 0; e < 8; ++e) acc[a][b][e] = 0.0f;

    const int cr  = tid >> 1;
    const int cc0 = (tid & 1) << 6;
    const unsigned char* gA = xq + (mBase + cr) * (long)K + cc0;
    const unsigned char* gB = wq + (nBase + cr) * (long)K + cc0;

#if USE_ASYNC_LDS
    auto issue_tile = [&](int buf, int kb) {
        const long kOff = (long)kb * BLK;
        unsigned char* lA = &As[buf][cr * 128 + cc0];
        unsigned char* lB = &Bs[buf][cr * 128 + cc0];
        #pragma unroll
        for (int i = 0; i < 4; ++i) {
            __builtin_amdgcn_global_load_async_to_lds_b128(
                as_g(gA + kOff + i * 16), as_l(lA + i * 16), 0, 0);
            __builtin_amdgcn_global_load_async_to_lds_b128(
                as_g(gB + kOff + i * 16), as_l(lB + i * 16), 0, 0);
        }
    };
#else
    auto issue_tile = [&](int buf, int kb) {
        const long kOff = (long)kb * BLK;
        unsigned char* lA = &As[buf][cr * 128 + cc0];
        unsigned char* lB = &Bs[buf][cr * 128 + cc0];
        #pragma unroll
        for (int i = 0; i < 4; ++i) {
            *(uint4*)(lA + i * 16) = *(const uint4*)(gA + kOff + i * 16);
            *(uint4*)(lB + i * 16) = *(const uint4*)(gB + kOff + i * 16);
        }
    };
#endif

    issue_tile(0, 0);
#if USE_ASYNC_LDS
    wait_async0();
#endif
    __syncthreads();

    for (int kb = 0; kb < J; ++kb) {
        const int cur = kb & 1;
        const unsigned char* Acur = As[cur];
        const unsigned char* Bcur = Bs[cur];

        if (kb + 1 < J) issue_tile(cur ^ 1, kb + 1);

        #pragma unroll
        for (int mt = 0; mt < 4; ++mt) {
            const int rowBase = waveM * 64 + mt * 16;

            v16i a;
            {
                const unsigned char* ap = Acur + (rowBase + lm) * 128 + hi * 16;
                #pragma unroll
                for (int p = 0; p < 8; ++p) {
                    int off = ((p >> 1) * 32) + ((p & 1) * 8);
                    int2 t = *(const int2*)(ap + off);
                    a[2 * p]     = t.x;
                    a[2 * p + 1] = t.y;
                }
            }

            const float* sp = &Ss[kb * 128 + rowBase + hi * 8];
            const float4 s0 = *(const float4*)(sp);
            const float4 s1 = *(const float4*)(sp + 4);
            const float sfac[8] = {s0.x, s0.y, s0.z, s0.w, s1.x, s1.y, s1.z, s1.w};

            #pragma unroll
            for (int nt = 0; nt < 2; ++nt) {
                const int colBase = waveN * 32 + nt * 16;
                v16i b;
                {
                    const unsigned char* bp = Bcur + (colBase + lm) * 128 + hi * 16;
                    #pragma unroll
                    for (int g = 0; g < 4; ++g) {
                        int4 t = *(const int4*)(bp + g * 32);
                        b[4 * g]     = t.x;
                        b[4 * g + 1] = t.y;
                        b[4 * g + 2] = t.z;
                        b[4 * g + 3] = t.w;
                    }
                }
                v8f czero = {};
                v8f part = __builtin_amdgcn_wmma_f32_16x16x128_fp8_fp8(
                    a, b, (short)0, czero, false, false);
                #pragma unroll
                for (int e = 0; e < 8; ++e)
                    acc[mt][nt][e] += part[e] * sfac[e];
            }
        }

#if USE_ASYNC_LDS
        wait_async0();
#endif
        __syncthreads();
    }

    #pragma unroll
    for (int mt = 0; mt < 4; ++mt) {
        #pragma unroll
        for (int nt = 0; nt < 2; ++nt) {
            const int col = (int)nBase + waveN * 32 + nt * 16 + lm;
            const float bv = bias[col];
            #pragma unroll
            for (int e = 0; e < 8; ++e) {
                long row = mBase + waveM * 64 + mt * 16 + e + hi * 8;
                out[row * (long)N + col] = acc[mt][nt][e] + bv;
            }
        }
    }
}

extern "C" void kernel_launch(void* const* d_in, const int* in_sizes, int n_in,
                              void* d_out, int out_size, void* d_ws, size_t ws_size,
                              hipStream_t stream) {
    const float* x      = (const float*)d_in[0];
    const float* w      = (const float*)d_in[1];
    const float* wscale = (const float*)d_in[2];
    const float* bias   = (const float*)d_in[3];
    float* out = (float*)d_out;

    const int O = in_sizes[3];
    const int H = in_sizes[1] / O;
    const int M = in_sizes[0] / H;
    const int J = H / BLK;

    unsigned char* xq  = (unsigned char*)d_ws;
    unsigned char* wqp = xq + (size_t)M * H;
    float*         xs  = (float*)(wqp + (size_t)O * H);

    {
        int groups = M * J;
        int blocks = (groups + 7) / 8;
        quantize_x_kernel<<<blocks, 256, 0, stream>>>(x, xq, xs, groups);
    }
    {
        long n4 = (long)O * H / 4;
        quantize_w_kernel<<<(int)((n4 + 255) / 256), 256, 0, stream>>>(w, wqp, n4);
    }
    {
        dim3 grid(O / 128, M / 128);
        fp8_gemm_kernel<<<grid, 256, 0, stream>>>(xq, wqp, xs, wscale, bias, out,
                                                  M, O, H);
    }
}
